// OuterProductMean_35089882808848
// MI455X (gfx1250) — matrix operand order repaired and checked by reading, not compiled, not run on hardware
//
#include <hip/hip_runtime.h>

typedef _Float16 v16h __attribute__((ext_vector_type(16)));
typedef _Float16 h8   __attribute__((ext_vector_type(8)));
typedef float    v8f  __attribute__((ext_vector_type(8)));
typedef float    f4   __attribute__((ext_vector_type(4)));

#define C_M 256
#define C_H 32
#define C_Z 128
#define DZ  (C_H * C_Z)   // 4096 flattened (d,z)

// Load one 16x32 f16 A-fragment row for v_wmma_f32_16x16x32_f16.
// ISA 7.12.2 (16-bit A, 16x32): lanes 0-15 hold K={0..7,16..23}, lanes 16-31
// hold K={8..15,24..31}, row M = lane&15. Per lane: two contiguous 8-f16
// chunks at K offsets half*8 and half*8+16 -> two aligned 16B loads.
__device__ __forceinline__ v16h load_a_frag(const _Float16* __restrict__ row32, int half) {
    h8 lo = *(const h8*)(row32 + half * 8);
    h8 hi = *(const h8*)(row32 + half * 8 + 16);
    v16h A;
#pragma unroll
    for (int e = 0; e < 8; ++e) { A[e] = lo[e]; A[e + 8] = hi[e]; }
    return A;
}

// ---------------------------------------------------------------------------
// Phase 1: LayerNorm over C_M + projections -> a16, b16 (f16, mask applied)
// One 256-thread block per row i.
// ---------------------------------------------------------------------------
__global__ void __launch_bounds__(C_M)
ln_proj(const float* __restrict__ m, const float* __restrict__ mask,
        const float* __restrict__ g, const float* __restrict__ beta,
        const float* __restrict__ w1, const float* __restrict__ b1,
        const float* __restrict__ w2, const float* __restrict__ b2,
        _Float16* __restrict__ a16, _Float16* __restrict__ b16)
{
    const int i = blockIdx.x;
    const int t = threadIdx.x;   // 0..255 == channel c

    __shared__ float s_ln[C_M];
    __shared__ float s_red[C_M];
    __shared__ float s_part[8];

    float x = m[(size_t)i * C_M + t];

    // mean (wave32 shuffle reduce + 8 partials in LDS)
    float v = x;
#pragma unroll
    for (int off = 16; off > 0; off >>= 1) v += __shfl_xor(v, off, 32);
    if ((t & 31) == 0) s_part[t >> 5] = v;
    __syncthreads();
    float mu = 0.f;
#pragma unroll
    for (int k = 0; k < 8; ++k) mu += s_part[k];
    mu *= (1.0f / C_M);
    __syncthreads();

    // variance
    float d = x - mu;
    float sq = d * d;
#pragma unroll
    for (int off = 16; off > 0; off >>= 1) sq += __shfl_xor(sq, off, 32);
    if ((t & 31) == 0) s_part[t >> 5] = sq;
    __syncthreads();
    float var = 0.f;
#pragma unroll
    for (int k = 0; k < 8; ++k) var += s_part[k];
    var *= (1.0f / C_M);

    s_ln[t] = d * rsqrtf(var + 1e-5f) * g[t] + beta[t];
    __syncthreads();

    // projections: thread t handles output h = t&31 over c-segment t>>5 (32 c)
    const int h = t & 31, seg = t >> 5;
    float pa = 0.f, pb = 0.f;
#pragma unroll
    for (int cc = 0; cc < 32; ++cc) {
        const int c = seg * 32 + cc;
        const float lv = s_ln[c];
        pa = fmaf(lv, w1[c * C_H + h], pa);
        pb = fmaf(lv, w2[c * C_H + h], pb);
    }
    const float mi = mask[i];

    s_red[t] = pa;
    __syncthreads();
    if (t < 32) {
        float s = b1[t];
#pragma unroll
        for (int k = 0; k < 8; ++k) s += s_red[k * 32 + t];
        a16[(size_t)i * C_H + t] = (_Float16)(s * mi);
    }
    __syncthreads();
    s_red[t] = pb;
    __syncthreads();
    if (t < 32) {
        float s = b2[t];
#pragma unroll
        for (int k = 0; k < 8; ++k) s += s_red[k * 32 + t];
        b16[(size_t)i * C_H + t] = (_Float16)(s * mi);
    }
}

// ---------------------------------------------------------------------------
// Phase 2: T[i, d, z] = sum_c a[i,c] * w_out[c,d,z]  via WMMA
// GEMM:  A = a16 [N x 32],  B = w_out viewed as [32 x 4096] (n = d*128 + z).
// K = 32 -> exactly one v_wmma_f32_16x16x32_f16 per 16x16 tile.
// Result stored TRANSPOSED as T16[i][z][d] (f16) so phase-3 B-frag loads are
// contiguous.
// ---------------------------------------------------------------------------
__global__ void __launch_bounds__(256)
make_T(const _Float16* __restrict__ a16,
       const float* __restrict__ w_out,
       _Float16* __restrict__ T16)
{
    const int wave = threadIdx.x >> 5;
    const int lane = threadIdx.x & 31;
    const int half = lane >> 4, lr = lane & 15;

    const int tile = blockIdx.x * 8 + wave;  // tile = it*256 + nt
    const int it = tile >> 8;
    const int nt = tile & 255;

    v16h A = load_a_frag(a16 + (size_t)(it * 16 + lr) * C_H, half);

    // B-frag: col n = lane&15, K = half*8 + (e&7) + (e>>3)*16 (matches A)
    const int n = nt * 16 + lr;
    v16h B;
#pragma unroll
    for (int e = 0; e < 16; ++e)
        B[e] = (_Float16)w_out[(size_t)(half * 8 + (e & 7) + (e >> 3) * 16) * DZ + n];

    v8f c = {};
    c = __builtin_amdgcn_wmma_f32_16x16x32_f16(false, A, false, B, (short)0, c,
                                               false, false);

    // C layout: VGPR r -> row = r + half*8, col = lane&15
    const int dcol = n >> 7;     // d
    const int z    = n & 127;    // z
#pragma unroll
    for (int r = 0; r < 8; ++r) {
        const int irow = it * 16 + r + half * 8;
        T16[(size_t)irow * DZ + z * C_H + dcol] = (_Float16)c[r];
    }
}

// ---------------------------------------------------------------------------
// Phase 3: O[i, j, z] = (sum_d b[j,d] * T[i,d,z] + b_out[z]) / (mask[i]^2+eps)
// Block = one i, 8 waves cover z=0..127 (one 16-wide z-tile each), loop over
// 4 j-tiles reusing the B fragment (T_i tile) held in VGPRs.
// Store-bound (512 MB output, written once, never re-read) -> LDS staging for
// fully coalesced 128-bit rows, written NON-TEMPORAL so the output stream
// does not evict the L2-resident T16 (8 MB, 16x reuse) / b16 (64 KB, 1024x).
// ---------------------------------------------------------------------------
__global__ void __launch_bounds__(256)
outer_gemm(const _Float16* __restrict__ b16,
           const _Float16* __restrict__ T16,
           const float* __restrict__ mask,
           const float* __restrict__ b_out,
           float* __restrict__ O, int N)
{
    const int i    = blockIdx.y;
    const int zt   = threadIdx.x >> 5;        // wave id = z tile
    const int lane = threadIdx.x & 31;
    const int half = lane >> 4, lr = lane & 15;

    // B-frag: col z = zt*16 + (lane&15); K rows d = half*16 + e -> contiguous
    // 16 f16 (32 B, aligned) thanks to transposed T16[i][z][d] layout.
    const int z = zt * 16 + lr;
    v16h B = *(const v16h*)(T16 + (size_t)i * DZ + (size_t)z * C_H + half * 16);

    const float mi  = mask[i];
    const float inv = 1.0f / (mi * mi + 0.001f);
    const float bo  = b_out[z];

    __shared__ __align__(16) float stage[16 * 132];   // pitch 132: bank-safe

#pragma unroll 1
    for (int q = 0; q < 4; ++q) {
        const int jt = blockIdx.x * 4 + q;

        v16h A = *(const v16h*)(b16 + (size_t)(jt * 16 + lr) * C_H + half * 16);

        v8f c = {};
        c = __builtin_amdgcn_wmma_f32_16x16x32_f16(false, A, false, B, (short)0,
                                                   c, false, false);

        // epilogue into LDS: row j_local = r + half*8, col = z
#pragma unroll
        for (int r = 0; r < 8; ++r) {
            const int jl = r + half * 8;
            stage[jl * 132 + z] = (c[r] + bo) * inv;
        }
        __syncthreads();

        // coalesced write-out: 16 rows x 128 f32 as non-temporal 128-bit stores
#pragma unroll
        for (int k = 0; k < 2; ++k) {
            const int idx = threadIdx.x + k * 256;
            const int row = idx >> 5;            // 0..15
            const int col = (idx & 31) << 2;     // 0,4,...,124
            f4 val = *(const f4*)&stage[row * 132 + col];
            f4* dst = (f4*)&O[(((size_t)i * N) + (size_t)jt * 16 + row) * C_Z + col];
            __builtin_nontemporal_store(val, dst);
        }
        __syncthreads();
    }
}

// ---------------------------------------------------------------------------
extern "C" void kernel_launch(void* const* d_in, const int* in_sizes, int n_in,
                              void* d_out, int out_size, void* d_ws, size_t ws_size,
                              hipStream_t stream)
{
    const float* m     = (const float*)d_in[0];
    const float* mask  = (const float*)d_in[1];
    const float* ln_g  = (const float*)d_in[2];
    const float* ln_b  = (const float*)d_in[3];
    const float* w1    = (const float*)d_in[4];
    const float* b1    = (const float*)d_in[5];
    const float* w2    = (const float*)d_in[6];
    const float* b2    = (const float*)d_in[7];
    const float* w_out = (const float*)d_in[8];
    const float* b_out = (const float*)d_in[9];
    float* O = (float*)d_out;

    const int N = in_sizes[0] / C_M;   // 1024

    // workspace layout: a16 | b16 | T16  (all f16)
    char* ws = (char*)d_ws;
    _Float16* a16 = (_Float16*)ws;
    _Float16* b16 = (_Float16*)(ws + (size_t)N * C_H * sizeof(_Float16));
    _Float16* T16 = (_Float16*)(ws + 2 * (size_t)N * C_H * sizeof(_Float16));

    ln_proj<<<N, C_M, 0, stream>>>(m, mask, ln_g, ln_b, w1, b1, w2, b2, a16, b16);
    // tiles = (N/16) * 256, 8 waves per block -> N*2 blocks
    make_T<<<N * 2, 256, 0, stream>>>(a16, w_out, T16);
    // grid.x covers j-tiles in groups of 4, grid.y = i
    outer_gemm<<<dim3(N / 64, N), 256, 0, stream>>>(b16, T16, mask, b_out, O, N);
}
